// GATCONV_48361331753435
// MI455X (gfx1250) — matrix operand order repaired and checked by reading, not compiled, not run on hardware
//
#include <hip/hip_runtime.h>

// ---------------- problem constants (match reference) ----------------
#define N_NODES 50000
#define N_EDGES 800000
#define EA      (N_EDGES + N_NODES)   // edges + self loops = 850000
#define F       128                   // feature width (F_IN == HC)
#define HEADS   4
#define DIM     32
#define HC      128
#define LAYERS  3
#define DIM2    16
#define NCLS    6
#define SLOPE   0.2f

// ---------------- WMMA types ----------------
typedef __attribute__((ext_vector_type(16))) __bf16 v16bf;
typedef __attribute__((ext_vector_type(8)))  float  v8f;

// monotone float <-> uint mapping for atomicMax on floats
__device__ __forceinline__ unsigned f2ord(float f) {
  unsigned u = __float_as_uint(f);
  return (u >> 31) ? ~u : (u | 0x80000000u);
}
__device__ __forceinline__ float ord2f(unsigned o) {
  return (o >> 31) ? __uint_as_float(o & 0x7fffffffu) : __uint_as_float(~o);
}

// ============================================================================
// Node transform: out[N,128] = h[N,128] @ W[128,128] + bias, via bf16 WMMA.
// Block = 256 threads = 8 waves; wave w handles 16-row tile (blockIdx.x*8+w)
// across all 8 column tiles. W^T staged bf16 in LDS (32KB).
// Native (__bf16) casts -> hardware v_cvt bf16 conversion (no SW rounding).
// ============================================================================
__global__ __launch_bounds__(256) void node_xform_wmma(
    const float* __restrict__ h, const float* __restrict__ W,
    const float* __restrict__ bias, float* __restrict__ out)
{
  __shared__ __align__(32) __bf16 wt[F * F];          // wt[n][k] = bf16(W[k][n])
  const int t = threadIdx.x;
  for (int i = t; i < F * F; i += 256) {
    int k = i >> 7, n = i & 127;                      // coalesced global read
    wt[n * F + k] = (__bf16)W[i];
  }
  __syncthreads();

  const int lane = t & 31;
  const int tile = blockIdx.x * 8 + (t >> 5);
  if (tile * 16 >= N_NODES) return;                   // uniform per-wave exit
  const int rowBase = tile * 16;
  const int lrow = lane & 15;
  const bool hi = lane >= 16;

  v8f acc[8] = {};                                    // 8 col tiles x 8 f32

  const float* hrow = h + (rowBase + lrow) * F;
  const int kbA = hi ? 16 : 0;   // A: lanes 0-15 -> K 0..15, 16-31 -> K 16..31
  const int kbB = hi ? 16 : 0;   // B: lanes 0-15 -> K 0..15, 16-31 -> K 16..31

  for (int kt = 0; kt < 4; ++kt) {
    const int k0 = kt * 32 + kbA;
    const int k1 = k0 + 8;
    float r[16];                                      // 4x global_load_b128
    *(float4*)&r[0]  = *(const float4*)(hrow + k0);
    *(float4*)&r[4]  = *(const float4*)(hrow + k0 + 4);
    *(float4*)&r[8]  = *(const float4*)(hrow + k1);
    *(float4*)&r[12] = *(const float4*)(hrow + k1 + 4);
    v16bf a;
#pragma unroll
    for (int j = 0; j < 16; ++j) a[j] = (__bf16)r[j]; // hw cvt, packed

#pragma unroll
    for (int nt = 0; nt < 8; ++nt) {
      const int col = nt * 16 + lrow;
      const v16bf b = *(const v16bf*)&wt[col * F + kt * 32 + kbB];
      acc[nt] = __builtin_amdgcn_wmma_f32_16x16x32_bf16(
          false, a, false, b, (short)0, acc[nt], false, false);
    }
  }

#pragma unroll
  for (int nt = 0; nt < 8; ++nt) {
    const int col = nt * 16 + lrow;
    const float bc = bias[col];
#pragma unroll
    for (int r = 0; r < 8; ++r) {                     // C layout: VGPR r -> M=r / M=r+8
      const int row = rowBase + r + (hi ? 8 : 0);
      out[row * F + col] = acc[nt][r] + bc;
    }
  }
}

// ============================================================================
// Edge kernels: wave32 per edge, lane = channel within head.
// ============================================================================
__global__ __launch_bounds__(256) void edge_score_max(
    const float* __restrict__ xl, const float* __restrict__ xr,
    const long long* __restrict__ ei, const float* __restrict__ att,
    float* __restrict__ score, unsigned* __restrict__ mord)
{
  const int e = blockIdx.x * 8 + (threadIdx.x >> 5);
  if (e >= EA) return;
  const int lane = threadIdx.x & 31;
  int src, dst;
  if (e < N_EDGES) { src = (int)ei[e]; dst = (int)ei[N_EDGES + e]; }
  else             { src = dst = e - N_EDGES; }       // self loop
  const float* xls = xl + (long)src * F;
  const float* xrd = xr + (long)dst * F;
  __builtin_prefetch(xls, 0, 0);
  float p[HEADS];
#pragma unroll
  for (int hh = 0; hh < HEADS; ++hh) {
    float m = xls[hh * DIM + lane] + xrd[hh * DIM + lane];
    float lr = m > 0.f ? m : SLOPE * m;
    p[hh] = lr * att[hh * DIM + lane];
  }
#pragma unroll
  for (int off = 16; off > 0; off >>= 1)
#pragma unroll
    for (int hh = 0; hh < HEADS; ++hh)
      p[hh] += __shfl_xor(p[hh], off, 32);
  if (lane == 0) {
#pragma unroll
    for (int hh = 0; hh < HEADS; ++hh) {
      score[e * HEADS + hh] = p[hh];
      atomicMax(&mord[dst * HEADS + hh], f2ord(p[hh]));
    }
  }
}

__global__ __launch_bounds__(256) void edge_expsum(
    const long long* __restrict__ ei, float* __restrict__ score,
    const unsigned* __restrict__ mord, float* __restrict__ ssum)
{
  const int idx = blockIdx.x * 256 + threadIdx.x;
  if (idx >= EA * HEADS) return;
  const int e = idx >> 2, hh = idx & 3;
  const int dst = (e < N_EDGES) ? (int)ei[N_EDGES + e] : e - N_EDGES;
  const float m = ord2f(mord[dst * HEADS + hh]);
  const float ex = __expf(score[idx] - m);
  score[idx] = ex;                                    // in-place: score -> exp
  atomicAdd(&ssum[dst * HEADS + hh], ex);
}

__global__ __launch_bounds__(256) void edge_aggregate(
    const float* __restrict__ xl, const long long* __restrict__ ei,
    const float* __restrict__ ex, const float* __restrict__ ssum,
    float* __restrict__ acc)
{
  const int e = blockIdx.x * 8 + (threadIdx.x >> 5);
  if (e >= EA) return;
  const int lane = threadIdx.x & 31;
  int src, dst;
  if (e < N_EDGES) { src = (int)ei[e]; dst = (int)ei[N_EDGES + e]; }
  else             { src = dst = e - N_EDGES; }
  const float* xls = xl + (long)src * F;
#pragma unroll
  for (int hh = 0; hh < HEADS; ++hh) {
    const float alpha = ex[e * HEADS + hh] / (ssum[dst * HEADS + hh] + 1e-16f);
    atomicAdd(&acc[dst * F + hh * DIM + lane], alpha * xls[hh * DIM + lane]);
  }
}

__global__ void layer_finish(const float* __restrict__ acc,
                             const float* __restrict__ cb,
                             float* __restrict__ hout)
{
  const int i = blockIdx.x * 256 + threadIdx.x;
  if (i >= N_NODES * F) return;
  const float v = acc[i] + cb[i & 127];
  hout[i] = v > 0.f ? v : 0.f;
}

__global__ void zero_u32(unsigned* __restrict__ p, int n) {
  const int i = blockIdx.x * 256 + threadIdx.x;
  if (i < n) p[i] = 0u;
}

// ============================================================================
// MLP head + edge output
// ============================================================================
__global__ void mlp1(const float* __restrict__ h, const float* __restrict__ W1,
                     const float* __restrict__ b1, float* __restrict__ h1) {
  const int idx = blockIdx.x * 256 + threadIdx.x;     // node*32 + c
  if (idx >= N_NODES * DIM) return;
  const int node = idx >> 5, c = idx & 31;
  const float* hr = h + node * F;
  float s = b1[c];
  for (int k = 0; k < F; ++k) s += hr[k] * W1[k * DIM + c];
  h1[idx] = s > 0.f ? s : 0.f;
}

__global__ void mlp2(const float* __restrict__ h1, const float* __restrict__ W2,
                     const float* __restrict__ b2, float* __restrict__ h2) {
  const int idx = blockIdx.x * 256 + threadIdx.x;     // node*16 + c
  if (idx >= N_NODES * DIM2) return;
  const int node = idx >> 4, c = idx & 15;
  const float* hr = h1 + node * DIM;
  float s = b2[c];
  for (int k = 0; k < DIM; ++k) s += hr[k] * W2[k * DIM2 + c];
  h2[idx] = s > 0.f ? s : 0.f;
}

__global__ void edge_head(const float* __restrict__ h2,
                          const long long* __restrict__ ei,
                          const float* __restrict__ W3, const float* __restrict__ b3,
                          float* __restrict__ out6, float* __restrict__ efeat) {
  const int e = blockIdx.x * 256 + threadIdx.x;
  if (e >= N_EDGES) return;
  const int src = (int)ei[e], dst = (int)ei[N_EDGES + e];
  float ef[2 * DIM2];
#pragma unroll
  for (int j = 0; j < DIM2; ++j) {
    ef[j]        = h2[src * DIM2 + j];
    ef[DIM2 + j] = h2[dst * DIM2 + j];
  }
#pragma unroll
  for (int j = 0; j < 2 * DIM2; ++j) efeat[(long)e * 32 + j] = ef[j];
#pragma unroll
  for (int c = 0; c < NCLS; ++c) {
    float s = b3[c];
#pragma unroll
    for (int k = 0; k < 2 * DIM2; ++k) s += ef[k] * W3[k * NCLS + c];
    out6[(long)e * NCLS + c] = s;
  }
}

// ============================================================================
extern "C" void kernel_launch(void* const* d_in, const int* in_sizes, int n_in,
                              void* d_out, int out_size, void* d_ws, size_t ws_size,
                              hipStream_t stream) {
  const float*     x   = (const float*)d_in[0];
  const long long* ei  = (const long long*)d_in[1];   // int64 edge_index [2,E]
  const float*     Wl  = (const float*)d_in[3];
  const float*     Wr  = (const float*)d_in[4];
  const float*     bl  = (const float*)d_in[5];
  const float*     br  = (const float*)d_in[6];
  const float*     att = (const float*)d_in[7];
  const float*     cb  = (const float*)d_in[8];
  const float*     W1  = (const float*)d_in[9];
  const float*     b1  = (const float*)d_in[10];
  const float*     W2  = (const float*)d_in[11];
  const float*     b2  = (const float*)d_in[12];
  const float*     W3  = (const float*)d_in[13];
  const float*     b3  = (const float*)d_in[14];

  // ---- workspace layout (floats) ----
  float* w = (float*)d_ws;
  float* hA    = w;                  w += (size_t)N_NODES * F;
  float* hB    = w;                  w += (size_t)N_NODES * F;
  float* xl    = w;                  w += (size_t)N_NODES * F;
  float* xr    = w;                  w += (size_t)N_NODES * F;
  float* acc   = w;                  w += (size_t)N_NODES * F;      // \ zeroed
  float* ssum  = w;                  w += (size_t)N_NODES * HEADS;  //  | as one
  unsigned* mord = (unsigned*)w;     w += (size_t)N_NODES * HEADS;  // / region
  float* score = w;                  w += (size_t)EA * HEADS;
  float* h1    = w;                  w += (size_t)N_NODES * DIM;
  float* h2s   = w;                  w += (size_t)N_NODES * DIM2;

  float* out6  = (float*)d_out;                       // [E, 6]
  float* efeat = out6 + (size_t)N_EDGES * NCLS;       // [E, 32]

  const int TILES   = (N_NODES + 15) / 16;            // 3125
  const int G_GEMM  = (TILES + 7) / 8;                // 391 blocks of 8 waves
  const int G_EDGE  = (EA + 7) / 8;                   // wave per edge
  const int G_EH    = (EA * HEADS + 255) / 256;
  const int G_NF    = (N_NODES * F + 255) / 256;
  const int G_ZERO  = (N_NODES * (F + 2 * HEADS) + 255) / 256;

  const float* hin = x;
  float* hout = hA;
  for (int l = 0; l < LAYERS; ++l) {
    const float* Wl_l = Wl + (size_t)l * F * HC;
    const float* Wr_l = Wr + (size_t)l * F * HC;
    node_xform_wmma<<<G_GEMM, 256, 0, stream>>>(hin, Wl_l, bl + l * HC, xl);
    node_xform_wmma<<<G_GEMM, 256, 0, stream>>>(hin, Wr_l, br + l * HC, xr);
    zero_u32<<<G_ZERO, 256, 0, stream>>>((unsigned*)acc,
                                         N_NODES * (F + 2 * HEADS));
    edge_score_max<<<G_EDGE, 256, 0, stream>>>(xl, xr, ei,
                                               att + l * HEADS * DIM, score, mord);
    edge_expsum<<<G_EH, 256, 0, stream>>>(ei, score, mord, ssum);
    edge_aggregate<<<G_EDGE, 256, 0, stream>>>(xl, ei, score, ssum, acc);
    layer_finish<<<G_NF, 256, 0, stream>>>(acc, cb + l * HC, hout);
    hin  = hout;
    hout = (hout == hA) ? hB : hA;
  }

  mlp1<<<(N_NODES * DIM  + 255) / 256, 256, 0, stream>>>(hin, W1, b1, h1);
  mlp2<<<(N_NODES * DIM2 + 255) / 256, 256, 0, stream>>>(h1, W2, b2, h2s);
  edge_head<<<(N_EDGES + 255) / 256, 256, 0, stream>>>(h2s, ei, W3, b3,
                                                       out6, efeat);
}
